// Attention_7481833029925
// MI455X (gfx1250) — matrix operand order repaired and checked by reading, not compiled, not run on hardware
//
#include <hip/hip_runtime.h>

// ---------------- problem constants ----------------
#define B_SZ     4
#define N_SEQ    2048
#define D_MODEL  512
#define HEADS    8
#define DIM_HEAD 64
#define SCALE_F  0.125f   // 64^-0.5

// ---------------- CDNA5 feature detection (compile-time safe) ----------------
#if __has_builtin(__builtin_amdgcn_global_load_async_to_lds_b128)
#  define HAVE_ASYNC_COPY 1
#else
#  define HAVE_ASYNC_COPY 0
#endif

#if HAVE_ASYNC_COPY
// Signature learned from hipcc diagnostics: param0 is a non-const pointer to
// 'int __vector(4)' in the global (__device__ / AS1) address space.
typedef int v4i_b128 __attribute__((vector_size(16)));
typedef __attribute__((address_space(1))) v4i_b128* as1_b128_t;
typedef __attribute__((address_space(3))) v4i_b128* as3_b128_t;

__device__ __forceinline__ void async_copy_b128(const unsigned short* g,
                                                unsigned short* l) {
    __builtin_amdgcn_global_load_async_to_lds_b128(
        (as1_b128_t)(void*)g, (as3_b128_t)(void*)l, 0, 0);
}
#endif

__device__ __forceinline__ void wait_async0() {
#if __has_builtin(__builtin_amdgcn_s_wait_asynccnt)
    __builtin_amdgcn_s_wait_asynccnt(0);
#else
    asm volatile("s_wait_asynccnt 0" ::: "memory");
#endif
}

// ---------------- WMMA fragment types (wave32, gfx1250) ----------------
typedef __attribute__((ext_vector_type(16))) __bf16 v16bf;
typedef __attribute__((ext_vector_type(8)))  float  v8f;

union Frag {
    v16bf          v;
    unsigned short s[16];
    unsigned int   u[8];
};

__device__ __forceinline__ unsigned short f2bf(float f) {
    unsigned int u = __float_as_uint(f);
    u += 0x7FFFu + ((u >> 16) & 1u);   // round-to-nearest-even
    return (unsigned short)(u >> 16);
}

// 16 contiguous bf16 (B-fragment run), 16B-aligned source
__device__ __forceinline__ void load_frag16(Frag& f, const unsigned short* p) {
    const uint4* q = (const uint4*)p;
    uint4 a = q[0], b = q[1];
    f.u[0] = a.x; f.u[1] = a.y; f.u[2] = a.z; f.u[3] = a.w;
    f.u[4] = b.x; f.u[5] = b.y; f.u[6] = b.z; f.u[7] = b.w;
}

// two runs of 8 contiguous bf16 (A-fragment), each 16B-aligned
__device__ __forceinline__ void load_frag_2x8(Frag& f, const unsigned short* lo,
                                              const unsigned short* hi) {
    uint4 a = *(const uint4*)lo;
    uint4 b = *(const uint4*)hi;
    f.u[0] = a.x; f.u[1] = a.y; f.u[2] = a.z; f.u[3] = a.w;
    f.u[4] = b.x; f.u[5] = b.y; f.u[6] = b.z; f.u[7] = b.w;
}

__device__ __forceinline__ v8f wmma_bf16(const Frag& a, const Frag& b, v8f c) {
    return __builtin_amdgcn_wmma_f32_16x16x32_bf16(false, a.v, false, b.v,
                                                   (short)0, c, false, false);
}

// reductions across the 16-lane half-groups of a wave32
__device__ __forceinline__ float redmax16(float x) {
#pragma unroll
    for (int off = 1; off < 16; off <<= 1) x = fmaxf(x, __shfl_xor(x, off, 32));
    return x;
}
__device__ __forceinline__ float redsum16(float x) {
#pragma unroll
    for (int off = 1; off < 16; off <<= 1) x += __shfl_xor(x, off, 32);
    return x;
}

// ---------------- kernel 0: one-shot f32 -> bf16 conversion of X ----------------
__global__ __launch_bounds__(256) void xcvt_kernel(const float* __restrict__ X,
                                                   unsigned short* __restrict__ Xb) {
    const size_t i = ((size_t)blockIdx.x * 256 + threadIdx.x) * 8;
    float4 a = *(const float4*)(X + i);
    float4 b = *(const float4*)(X + i + 4);
    union { uint4 q; unsigned short s[8]; } o;
    o.s[0] = f2bf(a.x); o.s[1] = f2bf(a.y); o.s[2] = f2bf(a.z); o.s[3] = f2bf(a.w);
    o.s[4] = f2bf(b.x); o.s[5] = f2bf(b.y); o.s[6] = f2bf(b.z); o.s[7] = f2bf(b.w);
    *(uint4*)(Xb + i) = o.q;
}

// ---------------- kernel 1: weight transpose + bf16 convert ----------------
// W (f32, [in=512][out=512]) -> T (bf16 raw, [out=512][in=512])
__global__ __launch_bounds__(256) void wtrans_kernel(
    const float* __restrict__ W0, const float* __restrict__ W1,
    const float* __restrict__ W2, const float* __restrict__ W3,
    unsigned short* __restrict__ T0, unsigned short* __restrict__ T1,
    unsigned short* __restrict__ T2, unsigned short* __restrict__ T3)
{
    __shared__ unsigned short tile[16][17];
    const float* W = W0; unsigned short* T = T0;
    if (blockIdx.z == 1)      { W = W1; T = T1; }
    else if (blockIdx.z == 2) { W = W2; T = T2; }
    else if (blockIdx.z == 3) { W = W3; T = T3; }

    const int tx = threadIdx.x, ty = threadIdx.y;
    const int o0 = blockIdx.x * 16, i0 = blockIdx.y * 16;
    tile[ty][tx] = f2bf(W[(size_t)(i0 + ty) * D_MODEL + o0 + tx]);
    __syncthreads();
    T[(size_t)(o0 + ty) * D_MODEL + i0 + tx] = tile[tx][ty];
}

// ---------------- kernel 2: fused QKV projection ----------------
// Xb (bf16 [8192][512]) @ Wt (bf16 [512 out][512 in]) -> Q/K/V bf16 [B,H,N,64]
__global__ __launch_bounds__(128) void qkv_proj_kernel(
    const unsigned short* __restrict__ Xb,
    const unsigned short* __restrict__ WTq,
    const unsigned short* __restrict__ WTk,
    const unsigned short* __restrict__ WTv,
    unsigned short* __restrict__ Qb,
    unsigned short* __restrict__ Kb,
    unsigned short* __restrict__ Vb)
{
    const unsigned short* Wt  = WTq;
    unsigned short*       Out = Qb;
    if (blockIdx.z == 1)      { Wt = WTk; Out = Kb; }
    else if (blockIdx.z == 2) { Wt = WTv; Out = Vb; }

    const int wave = threadIdx.x >> 5;
    const int lane = threadIdx.x & 31;
    const int half = lane >> 4;
    const int ln   = lane & 15;
    const int m0   = (blockIdx.x * 4 + wave) * 16;   // row tile in [0,8192)
    const int n0   = blockIdx.y * 16;                // col tile in [0,512)

    v8f c = {0.f, 0.f, 0.f, 0.f, 0.f, 0.f, 0.f, 0.f};
#pragma unroll 4
    for (int kc = 0; kc < 16; ++kc) {
        Frag a, bf;
        const unsigned short* arow = Xb + (size_t)(m0 + ln) * D_MODEL + kc * 32;
        load_frag_2x8(a, arow + half * 16, arow + half * 16 + 8);
        load_frag16(bf, Wt + (size_t)(n0 + ln) * D_MODEL + kc * 32 + half * 16);
        c = wmma_bf16(a, bf, c);
    }

    const int h = n0 >> 6;
    const int d = (n0 & 63) + ln;
#pragma unroll
    for (int r = 0; r < 8; ++r) {
        const int row = m0 + half * 8 + r;
        const int bb  = row >> 11;          // / N_SEQ
        const int tok = row & (N_SEQ - 1);
        Out[((size_t)(bb * HEADS + h) * N_SEQ + tok) * DIM_HEAD + d] = f2bf(c[r]);
    }
}

// ---------------- kernel 3: flash attention ----------------
// block = 4 waves; wave handles 16 query rows; K/V streamed in 32-key chunks via LDS;
// K chunk uses CDNA5 async global->LDS copy (ASYNCcnt) when available.
__global__ __launch_bounds__(128) void attn_kernel(
    const unsigned short* __restrict__ Qb,
    const unsigned short* __restrict__ Kb,
    const unsigned short* __restrict__ Vb,
    const float* __restrict__ pb,          // [H][N][N]
    unsigned short* __restrict__ AO)       // bf16 [B*N][512]
{
    __shared__ __align__(16) unsigned short sK[32 * 80];      // [key][dim], stride 80
    __shared__ __align__(16) unsigned short sV[64 * 40];      // [dim][key] transposed
    __shared__ __align__(16) unsigned short sP[4][16 * 40];   // per-wave P tile

    const int tid  = threadIdx.x;
    const int wave = tid >> 5;
    const int lane = tid & 31;
    const int half = lane >> 4;
    const int ln   = lane & 15;

    const int bh = blockIdx.y;             // b*HEADS + h
    const int b  = bh >> 3;
    const int h  = bh & 7;
    const int q0 = blockIdx.x * 64 + wave * 16;

    const unsigned short* Qh = Qb + (size_t)bh * N_SEQ * DIM_HEAD;
    const unsigned short* Kh = Kb + (size_t)bh * N_SEQ * DIM_HEAD;
    const unsigned short* Vh = Vb + (size_t)bh * N_SEQ * DIM_HEAD;
    const float* bias = pb + ((size_t)h * N_SEQ + q0) * N_SEQ;

    // Q A-fragments (head-dim split into two K=32 chunks), reused for all key chunks
    Frag aq0, aq1;
    {
        const unsigned short* qrow = Qh + (size_t)(q0 + ln) * DIM_HEAD;
        load_frag_2x8(aq0, qrow + half * 16,      qrow + 8 + half * 16);
        load_frag_2x8(aq1, qrow + 32 + half * 16, qrow + 40 + half * 16);
    }

    float rm[8], rl[8];
    v8f o0 = {0.f,0.f,0.f,0.f,0.f,0.f,0.f,0.f};
    v8f o1 = o0, o2 = o0, o3 = o0;
#pragma unroll
    for (int r = 0; r < 8; ++r) { rm[r] = -1e30f; rl[r] = 0.f; }

    const int ldr = tid >> 2;          // 0..31: key row of chunk this thread loads
    const int ldc = (tid & 3) * 16;    // dim column base

    for (int n0 = 0; n0 < N_SEQ; n0 += 32) {
        __syncthreads();               // previous chunk fully consumed

        // ---- K chunk: async copy straight into LDS (overlaps with V transpose) ----
        const unsigned short* ks = Kh + (size_t)(n0 + ldr) * DIM_HEAD + ldc;
#if HAVE_ASYNC_COPY
        async_copy_b128(ks,     &sK[ldr * 80 + ldc]);
        async_copy_b128(ks + 8, &sK[ldr * 80 + ldc + 8]);
#else
        {
            uint4 k0 = *(const uint4*)ks;
            uint4 k1 = *(const uint4*)(ks + 8);
            *(uint4*)&sK[ldr * 80 + ldc]     = k0;
            *(uint4*)&sK[ldr * 80 + ldc + 8] = k1;
        }
#endif
        // ---- V chunk: load + transpose-scatter into LDS ----
        {
            const unsigned short* vs = Vh + (size_t)(n0 + ldr) * DIM_HEAD + ldc;
            union { uint4 q[2]; unsigned short s[16]; } vv;
            vv.q[0] = *(const uint4*)vs;
            vv.q[1] = *(const uint4*)(vs + 8);
#pragma unroll
            for (int i = 0; i < 16; ++i) sV[(ldc + i) * 40 + ldr] = vv.s[i];
        }
        // prefetch next chunk while this one is consumed
        if (n0 + 32 < N_SEQ) {
            __builtin_prefetch(Kh + (size_t)(n0 + 32 + ldr) * DIM_HEAD + ldc, 0, 1);
            __builtin_prefetch(Vh + (size_t)(n0 + 32 + ldr) * DIM_HEAD + ldc, 0, 1);
        }
#if HAVE_ASYNC_COPY
        wait_async0();
#endif
        __syncthreads();

        // ---- scores: S = Q @ K^T  (two 16x16 tiles over the 32-key chunk) ----
        v8f s0 = {0.f,0.f,0.f,0.f,0.f,0.f,0.f,0.f};
        v8f s1 = s0;
        {
            Frag bk;
            load_frag16(bk, &sK[ln * 80 + half * 16]);             s0 = wmma_bf16(aq0, bk, s0);
            load_frag16(bk, &sK[ln * 80 + 32 + half * 16]);        s0 = wmma_bf16(aq1, bk, s0);
            load_frag16(bk, &sK[(16 + ln) * 80 + half * 16]);      s1 = wmma_bf16(aq0, bk, s1);
            load_frag16(bk, &sK[(16 + ln) * 80 + 32 + half * 16]); s1 = wmma_bf16(aq1, bk, s1);
        }

        // ---- scale + bias + online softmax ----
        float cmax[8];
#pragma unroll
        for (int r = 0; r < 8; ++r) {
            const size_t brow = (size_t)(half * 8 + r) * N_SEQ + n0 + ln;
            float x0 = s0[r] * SCALE_F + bias[brow];
            float x1 = s1[r] * SCALE_F + bias[brow + 16];
            s0[r] = x0; s1[r] = x1;
            cmax[r] = redmax16(fmaxf(x0, x1));
        }
        float alpha[8], csum[8];
#pragma unroll
        for (int r = 0; r < 8; ++r) {
            const float mnew = fmaxf(rm[r], cmax[r]);
            alpha[r] = __expf(rm[r] - mnew);
            rm[r] = mnew;
            const float p0 = __expf(s0[r] - mnew);
            const float p1 = __expf(s1[r] - mnew);
            s0[r] = p0; s1[r] = p1;
            csum[r] = redsum16(p0 + p1);
        }
#pragma unroll
        for (int r = 0; r < 8; ++r) {
            rl[r] = rl[r] * alpha[r] + csum[r];
            o0[r] *= alpha[r]; o1[r] *= alpha[r];
            o2[r] *= alpha[r]; o3[r] *= alpha[r];
        }

        // ---- P: C-layout -> A-layout via wave-private LDS tile ----
        unsigned short* pw = &sP[wave][0];
#pragma unroll
        for (int r = 0; r < 8; ++r) {
            pw[(half * 8 + r) * 40 + ln]      = f2bf(s0[r]);
            pw[(half * 8 + r) * 40 + 16 + ln] = f2bf(s1[r]);
        }
        asm volatile("" ::: "memory");   // LDS ops are in-order within a wave

        Frag ap;
        load_frag_2x8(ap, pw + ln * 40 + half * 16, pw + ln * 40 + 8 + half * 16);

        // ---- O += P @ V ----
        Frag bv;
        load_frag16(bv, &sV[ln * 40 + half * 16]);        o0 = wmma_bf16(ap, bv, o0);
        load_frag16(bv, &sV[(16 + ln) * 40 + half * 16]); o1 = wmma_bf16(ap, bv, o1);
        load_frag16(bv, &sV[(32 + ln) * 40 + half * 16]); o2 = wmma_bf16(ap, bv, o2);
        load_frag16(bv, &sV[(48 + ln) * 40 + half * 16]); o3 = wmma_bf16(ap, bv, o3);
    }

    // ---- normalize + write merged-head output [B*N][H*Dh] as bf16 ----
#pragma unroll
    for (int r = 0; r < 8; ++r) {
        const float inv = 1.0f / rl[r];
        const size_t base = (size_t)(b * N_SEQ + q0 + half * 8 + r) * (HEADS * DIM_HEAD)
                          + h * DIM_HEAD + ln;
        AO[base]      = f2bf(o0[r] * inv);
        AO[base + 16] = f2bf(o1[r] * inv);
        AO[base + 32] = f2bf(o2[r] * inv);
        AO[base + 48] = f2bf(o3[r] * inv);
    }
}

// ---------------- kernel 4: output projection ----------------
// AO (bf16 [8192][512]) @ WTo (bf16 [512][512] transposed) -> d_out f32
__global__ __launch_bounds__(128) void out_proj_kernel(
    const unsigned short* __restrict__ A,
    const unsigned short* __restrict__ WTo,
    float* __restrict__ Out)
{
    const int wave = threadIdx.x >> 5;
    const int lane = threadIdx.x & 31;
    const int half = lane >> 4;
    const int ln   = lane & 15;
    const int m0   = (blockIdx.x * 4 + wave) * 16;
    const int n0   = blockIdx.y * 16;

    v8f c = {0.f,0.f,0.f,0.f,0.f,0.f,0.f,0.f};
#pragma unroll 4
    for (int kc = 0; kc < 16; ++kc) {
        Frag a, bf;
        const unsigned short* arow = A + (size_t)(m0 + ln) * D_MODEL + kc * 32;
        load_frag_2x8(a, arow + half * 16, arow + half * 16 + 8);
        load_frag16(bf, WTo + (size_t)(n0 + ln) * D_MODEL + kc * 32 + half * 16);
        c = wmma_bf16(a, bf, c);
    }
#pragma unroll
    for (int r = 0; r < 8; ++r)
        Out[(size_t)(m0 + half * 8 + r) * D_MODEL + n0 + ln] = c[r];
}

// ---------------- host launch ----------------
extern "C" void kernel_launch(void* const* d_in, const int* in_sizes, int n_in,
                              void* d_out, int out_size, void* d_ws, size_t ws_size,
                              hipStream_t stream) {
    (void)in_sizes; (void)n_in; (void)out_size; (void)ws_size;

    const float* query = (const float*)d_in[0];
    const float* pbias = (const float*)d_in[1];
    const float* Wq    = (const float*)d_in[2];
    const float* Wk    = (const float*)d_in[3];
    const float* Wv    = (const float*)d_in[4];
    const float* Wo    = (const float*)d_in[5];

    char* ws = (char*)d_ws;
    const size_t WT_BYTES  = (size_t)D_MODEL * D_MODEL * 2;           // 512 KB each
    const size_t ACT_BYTES = (size_t)B_SZ * N_SEQ * D_MODEL * 2;      // 8 MB each
    unsigned short* WTq = (unsigned short*)(ws + 0 * WT_BYTES);
    unsigned short* WTk = (unsigned short*)(ws + 1 * WT_BYTES);
    unsigned short* WTv = (unsigned short*)(ws + 2 * WT_BYTES);
    unsigned short* WTo = (unsigned short*)(ws + 3 * WT_BYTES);
    char* act = ws + 4 * WT_BYTES;
    unsigned short* Xb = (unsigned short*)(act + 0 * ACT_BYTES);
    unsigned short* Qb = (unsigned short*)(act + 1 * ACT_BYTES);
    unsigned short* Kb = (unsigned short*)(act + 2 * ACT_BYTES);
    unsigned short* Vb = (unsigned short*)(act + 3 * ACT_BYTES);
    unsigned short* AO = (unsigned short*)(act + 4 * ACT_BYTES);

    const int xcvt_blocks = (B_SZ * N_SEQ * D_MODEL) / (256 * 8);     // 2048
    xcvt_kernel<<<dim3(xcvt_blocks), dim3(256), 0, stream>>>(query, Xb);

    wtrans_kernel<<<dim3(32, 32, 4), dim3(16, 16), 0, stream>>>(
        Wq, Wk, Wv, Wo, WTq, WTk, WTv, WTo);

    qkv_proj_kernel<<<dim3(128, 32, 3), dim3(128), 0, stream>>>(
        Xb, WTq, WTk, WTv, Qb, Kb, Vb);

    attn_kernel<<<dim3(N_SEQ / 64, B_SZ * HEADS), dim3(128), 0, stream>>>(
        Qb, Kb, Vb, pbias, AO);

    out_proj_kernel<<<dim3(128, 32), dim3(128), 0, stream>>>(
        AO, WTo, (float*)d_out);
}
